// PointLSTMCell_80934363726019
// MI455X (gfx1250) — matrix operand order repaired and checked by reading, not compiled, not run on hardware
//
#include <hip/hip_runtime.h>

typedef __attribute__((ext_vector_type(16))) _Float16 v16h;
typedef __attribute__((ext_vector_type(8)))  float    v8f;

#define B_     4
#define N_     1024
#define KN     32
#define CIN    132
#define HID    128
#define C1     264          // IN_CH + OFF + HID
#define K1P    288          // C1 padded to 9*32
#define C2     256          // 2*HID
#define ACT_STRIDE 296      // padded halves per column (16B aligned, conflict-free)
#define XS_STRIDE  264      // padded halves per column
#define COLS   128
#define PIX_PER_WG 4
#define BHN    (B_*HID*N_)  // 524288 elements per pooled output

#define W1_ELEMS (512*K1P)
#define W2_ELEMS (512*C2)

__device__ __forceinline__ float rcp_(float x){ return __builtin_amdgcn_rcpf(x); }
__device__ __forceinline__ float sigm(float x){ return rcp_(1.0f + __expf(-x)); }
__device__ __forceinline__ float logsig(float x){
    return fminf(x, 0.0f) - __logf(1.0f + __expf(-fabsf(x)));
}
__device__ __forceinline__ float tanh_(float x){
    float t = __expf(-2.0f*fabsf(x));
    float r = (1.0f - t) * rcp_(1.0f + t);
    return copysignf(r, x);
}

__device__ __forceinline__ v8f wmma_f16(v16h a, v16h b, v8f c){
    // (neg_a, A, neg_b, B, c_mod, C, reuse_a, reuse_b)
    return __builtin_amdgcn_wmma_f32_16x16x32_f16(false, a, false, b, (short)0, c, false, false);
}

// A fragment: 16x32 f16 weight tile, row-major [row][strideH] in global f16.
// lane holds row M = lane%16; lo half-lanes K blocks {0-7,16-23}, hi {8-15,24-31}.
__device__ __forceinline__ v16h ld_a_frag(const unsigned short* wbase,
                                          int row, int strideH, int kb, int hi){
    const unsigned short* p = wbase + (size_t)row*strideH + kb + (hi ? 8 : 0);
    union { v16h v; uint4 q[2]; } u;
    u.q[0] = *(const uint4*)(p);
    u.q[1] = *(const uint4*)(p + 16);
    return u.v;
}

// B fragment: 32x16 f16 activation tile, column-major in LDS [col][strideH].
// lane holds column = lane%16; lo half-lanes K blocks {0-7,16-23}, hi {8-15,24-31}.
__device__ __forceinline__ v16h ld_b_frag(const _Float16* lds, int col, int strideH,
                                          int kb, int hi){
    const _Float16* p = lds + col*strideH + kb + (hi ? 8 : 0);
    union { v16h v; uint4 q[2]; } u;
    u.q[0] = *(const uint4*)(p);
    u.q[1] = *(const uint4*)(p + 16);
    return u.v;
}

extern "C" __global__ void cvt_weights(const float* __restrict__ W1,
                                       const float* __restrict__ W2,
                                       unsigned short* __restrict__ ws){
    int i = blockIdx.x*256 + threadIdx.x;
    _Float16* o = (_Float16*)ws;
    if (i < W1_ELEMS) {
        int m = i / K1P, k = i % K1P;
        float v = (k < C1) ? W1[m*C1 + k] : 0.0f;
        o[i] = (_Float16)v;
    } else if (i < W1_ELEMS + W2_ELEMS) {
        o[i] = (_Float16)W2[i - W1_ELEMS];
    }
}

extern "C" __global__ __launch_bounds__(256)
void pointlstm_fused(const float* __restrict__ xin,  const float* __restrict__ hin,
                     const float* __restrict__ cellin, const float* __restrict__ normin,
                     const float* __restrict__ matchin,
                     const unsigned short* __restrict__ w1h,
                     const unsigned short* __restrict__ w2h,
                     const float* __restrict__ b1, const float* __restrict__ b2,
                     float* __restrict__ out)
{
    extern __shared__ __align__(16) _Float16 smem[];
    _Float16* act = smem;                      // [COLS][ACT_STRIDE] f16
    _Float16* xs  = smem + COLS*ACT_STRIDE;    // [COLS][XS_STRIDE]  f16
    float*    bsm = (float*)(xs + COLS*XS_STRIDE);  // [1024] biases (b1 | b2)

    const int tid  = threadIdx.x;
    const int lane = tid & 31;
    const int wv   = tid >> 5;        // wave id == M-tile (8 waves, 8 tiles of 16 rows)
    const int hi   = lane >> 4;       // lane half
    const int lcol = lane & 15;
    const int pixBase = blockIdx.x * PIX_PER_WG;

    // ---- stage combined = [input ; hidden(with offset-sub)] into LDS as f16 ----
    for (int idx = tid; idx < C1*COLS; idx += 256) {
        int c   = idx >> 7;
        int col = idx & 127;
        int pix = pixBase + (col >> 5);
        int b   = pix >> 10;
        int n   = pix & (N_ - 1);
        int kk  = col & 31;
        float v;
        if (c < CIN) {
            v = xin[(((size_t)b*CIN + c)*N_ + n)*KN + kk];
        } else {
            int c2 = c - CIN;
            v = hin[(((size_t)b*CIN + c2)*N_ + n)*KN + kk];
            if (c2 < 4) v -= xin[(((size_t)b*CIN + c2)*N_ + n)*KN + kk];
        }
        act[col*ACT_STRIDE + c] = (_Float16)v;
    }
    for (int idx = tid; idx < (K1P - C1)*COLS; idx += 256) {
        int c   = C1 + (idx >> 7);
        int col = idx & 127;
        act[col*ACT_STRIDE + c] = (_Float16)0.0f;
    }
    for (int i = tid; i < 1024; i += 256)
        bsm[i] = (i < 512) ? b1[i] : b2[i - 512];
    __syncthreads();

    const v8f zero8 = {0.f,0.f,0.f,0.f,0.f,0.f,0.f,0.f};

    #pragma unroll 1
    for (int pl = 0; pl < PIX_PER_WG; ++pl) {
        const int colbase = pl * 32;
        const int pix = pixBase + pl;
        const int b = pix >> 10;
        const int n = pix & (N_ - 1);

        // Opaque zero offset per pixel-iteration: blocks LICM/CSE of the weight
        // fragment loads across pixels (which previously spilled 288 VGPRs),
        // while keeping pointer provenance so loads stay global_load (not flat).
        int zoff = 0;
        asm volatile("" : "+v"(zoff));
        const unsigned short* w1p = w1h + zoff;
        const unsigned short* w2p = w2h + zoff;

        // ================= GEMM 1: gates = W_conv x combined =================
        v8f acc[4][2];
        #pragma unroll
        for (int g = 0; g < 4; ++g) { acc[g][0] = zero8; acc[g][1] = zero8; }

        #pragma unroll 1
        for (int ks = 0; ks < 9; ++ks) {
            const int kb = ks * 32;
            v16h af[4];
            #pragma unroll
            for (int g = 0; g < 4; ++g)
                af[g] = ld_a_frag(w1p, g*HID + (wv<<4) + lcol, K1P, kb, hi);
            v16h bf0 = ld_b_frag(act, colbase + lcol,      ACT_STRIDE, kb, hi);
            v16h bf1 = ld_b_frag(act, colbase + 16 + lcol, ACT_STRIDE, kb, hi);
            #pragma unroll
            for (int g = 0; g < 4; ++g) {
                acc[g][0] = wmma_f16(af[g], bf0, acc[g][0]);
                acc[g][1] = wmma_f16(af[g], bf1, acc[g][1]);
            }
        }

        // ---- epilogue 1: LSTM gate math, keep c_lstm in regs, xs -> LDS ----
        float cacc[2][8];
        #pragma unroll
        for (int nt = 0; nt < 2; ++nt) {
            const int col = colbase + nt*16 + lcol;
            const int kk  = nt*16 + lcol;
            #pragma unroll
            for (int j = 0; j < 8; ++j) {
                const int m = (wv<<4) + (hi<<3) + j;
                float xi = acc[0][nt][j] + bsm[m];
                float xf = acc[1][nt][j] + bsm[HID + m];
                float xo = acc[2][nt][j] + bsm[2*HID + m];
                float xg = acc[3][nt][j] + bsm[3*HID + m];
                float ig = sigm(xi), fg = sigm(xf), og = sigm(xo), gg = tanh_(xg);
                float cell = cellin[(((size_t)b*HID + m)*N_ + n)*KN + kk];
                float cl = fg*cell + ig*gg;
                float hl = og*tanh_(cl);
                cacc[nt][j] = cl;
                xs[col*XS_STRIDE + m]       = (_Float16)cl;
                xs[col*XS_STRIDE + HID + m] = (_Float16)hl;
            }
        }
        __syncthreads();

        // ================= GEMM 2: W_ifog x [c_lstm ; h_lstm] =================
        #pragma unroll
        for (int g = 0; g < 4; ++g) { acc[g][0] = zero8; acc[g][1] = zero8; }

        #pragma unroll 1
        for (int ks = 0; ks < 8; ++ks) {
            const int kb = ks * 32;
            v16h af[4];
            #pragma unroll
            for (int g = 0; g < 4; ++g)
                af[g] = ld_a_frag(w2p, g*HID + (wv<<4) + lcol, C2, kb, hi);
            v16h bf0 = ld_b_frag(xs, colbase + lcol,      XS_STRIDE, kb, hi);
            v16h bf1 = ld_b_frag(xs, colbase + 16 + lcol, XS_STRIDE, kb, hi);
            #pragma unroll
            for (int g = 0; g < 4; ++g) {
                acc[g][0] = wmma_f16(af[g], bf0, acc[g][0]);
                acc[g][1] = wmma_f16(af[g], bf1, acc[g][1]);
            }
        }

        // ---- epilogue 2: stabilized gates + next-state + max-pool over K ----
        float pm[4][8];   // running max of h_next, c_next, n_next, m_next
        #pragma unroll
        for (int q = 0; q < 4; ++q)
            #pragma unroll
            for (int j = 0; j < 8; ++j)
                pm[q][j] = -3.402823466e38f;

        #pragma unroll
        for (int nt = 0; nt < 2; ++nt) {
            const int kk = nt*16 + lcol;
            #pragma unroll
            for (int j = 0; j < 8; ++j) {
                const int m = (wv<<4) + (hi<<3) + j;
                float xi = acc[0][nt][j] + bsm[512 + m];
                float xf = acc[1][nt][j] + bsm[512 + HID + m];
                float xo = acc[2][nt][j] + bsm[512 + 2*HID + m];
                float xg = acc[3][nt][j] + bsm[512 + 3*HID + m];
                float oxs = sigm(xo);
                float gxs = tanh_(xg);
                float li  = logsig(xi);
                float lf  = logsig(xf);
                size_t gix = (((size_t)b*HID + m)*N_ + n)*KN + kk;
                float mt = matchin[gix];
                float nm = normin[gix];
                float mnext = fmaxf(lf + mt, li);
                float ist = sigm(li - mnext);
                float fst = sigm(lf + mt + mnext);
                float cnext = fst*cacc[nt][j] + ist*gxs;
                float nnext = fst*nm + ist;
                float hnext = oxs*cnext*rcp_(nnext);
                pm[0][j] = fmaxf(pm[0][j], hnext);
                pm[1][j] = fmaxf(pm[1][j], cnext);
                pm[2][j] = fmaxf(pm[2][j], nnext);
                pm[3][j] = fmaxf(pm[3][j], mnext);
            }
        }

        // cross-lane max over the 16 columns held in each lane half (wave32)
        #pragma unroll
        for (int q = 0; q < 4; ++q)
            #pragma unroll
            for (int j = 0; j < 8; ++j)
                #pragma unroll
                for (int off = 1; off < 16; off <<= 1)
                    pm[q][j] = fmaxf(pm[q][j], __shfl_xor(pm[q][j], off, 32));

        if (lcol == 0) {
            #pragma unroll
            for (int j = 0; j < 8; ++j) {
                const int m = (wv<<4) + (hi<<3) + j;
                size_t o = (size_t)(b*HID + m)*N_ + n;
                out[o]          = pm[0][j];   // pool(h_next)
                out[BHN   + o]  = pm[1][j];   // pool(c_next)
                out[2*BHN + o]  = pm[2][j];   // pool(n_next)
                out[3*BHN + o]  = pm[3][j];   // pool(m_next)
            }
        }
    }
}

extern "C" void kernel_launch(void* const* d_in, const int* in_sizes, int n_in,
                              void* d_out, int out_size, void* d_ws, size_t ws_size,
                              hipStream_t stream) {
    const float* xin    = (const float*)d_in[0];   // input_tensor  [4,132,1024,32]
    const float* hin    = (const float*)d_in[1];   // hidden_state  [4,132,1024,32]
    const float* cellin = (const float*)d_in[2];   // cell_state    [4,128,1024,32]
    const float* normin = (const float*)d_in[3];   // normal_state
    const float* matchin= (const float*)d_in[4];   // match_state
    const float* Wc     = (const float*)d_in[5];   // W_conv [512,264]
    const float* bc     = (const float*)d_in[6];   // b_conv [512]
    const float* Wi     = (const float*)d_in[7];   // W_ifog [512,256]
    const float* bi     = (const float*)d_in[8];   // b_ifog [512]

    unsigned short* ws = (unsigned short*)d_ws;    // f16 weights: W1 padded + W2
    const int tot = W1_ELEMS + W2_ELEMS;
    cvt_weights<<<(tot + 255)/256, 256, 0, stream>>>(Wc, Wi, ws);

    const size_t shbytes = (size_t)(COLS*ACT_STRIDE + COLS*XS_STRIDE) * 2 + 1024*4;
    pointlstm_fused<<<(B_*N_)/PIX_PER_WG, 256, shbytes, stream>>>(
        xin, hin, cellin, normin, matchin,
        ws, ws + W1_ELEMS, bc, bi, (float*)d_out);
}
